// BuiltControlled_31662498906409
// MI455X (gfx1250) — matrix operand order repaired and checked by reading, not compiled, not run on hardware
//
#include <hip/hip_runtime.h>
#include <hip/hip_bf16.h>

// Problem constants from the reference (fixed by setup_inputs).
#define BATCH    256
#define DIM      8192
#define KCH      32     // K per chunk = bf16 WMMA K
#define BN       32     // N columns per workgroup
#define NTHREADS 256    // 8 waves (wave32)

typedef __bf16 bf16_t;
typedef __attribute__((ext_vector_type(16))) __bf16 v16bf;
typedef __attribute__((ext_vector_type(8)))  float  v8f;

struct I8 { int4 lo, hi; };

static __device__ __forceinline__ v16bf make_frag(int4 lo, int4 hi) {
    I8 t{lo, hi};
    return __builtin_bit_cast(v16bf, t);
}

static __device__ __forceinline__ v8f wmma_bf16(v16bf a, v16bf b, v8f c) {
    // (neg_a, A, neg_b, B, c_mod, C, reuse_a, reuse_b)
    return __builtin_amdgcn_wmma_f32_16x16x32_bf16(false, a, false, b, (short)0, c,
                                                   false, false);
}

// ---------------------------------------------------------------------------
// Prep: split fp32 complex state into 4 bf16 planes (re_hi, re_lo, im_hi,
// im_lo) with the control-bit K-mask applied. 16 MB into d_ws.
// ---------------------------------------------------------------------------
__global__ __launch_bounds__(NTHREADS) void prep_a(
    const float2* __restrict__ state,
    const int* __restrict__ pctrl, const int* __restrict__ pnq,
    bf16_t* __restrict__ ws)
{
    const int control = pctrl[0];
    const int nq      = pnq[0];
    const int mask    = 1 << (nq - control - 1);
    const size_t P    = (size_t)BATCH * DIM;
    bf16_t* arh = ws;
    bf16_t* arl = ws + P;
    bf16_t* aih = ws + 2 * P;
    bf16_t* ail = ws + 3 * P;

    size_t base = ((size_t)blockIdx.x * NTHREADS + threadIdx.x) * 8;
#pragma unroll
    for (int j = 0; j < 8; ++j) {
        size_t idx = base + j;
        int k = (int)(idx & (DIM - 1));
        float re = 0.f, im = 0.f;
        if (k & mask) { float2 v = state[idx]; re = v.x; im = v.y; }
        bf16_t rh = (bf16_t)re;
        bf16_t rl = (bf16_t)(re - (float)rh);
        bf16_t ih = (bf16_t)im;
        bf16_t il = (bf16_t)(im - (float)ih);
        arh[idx] = rh; arl[idx] = rl; aih[idx] = ih; ail[idx] = il;
    }
}

// ---------------------------------------------------------------------------
// Controlled-unitary complex GEMM with split-bf16 fp32 emulation.
// Each block: all 256 batch rows x 32 output columns; streams B once.
// ---------------------------------------------------------------------------
__global__ __launch_bounds__(NTHREADS) void cgemm_ctrl(
    const float2* __restrict__ state, const float2* __restrict__ tmat,
    const int* __restrict__ pctrl, const int* __restrict__ pnq,
    const bf16_t* __restrict__ ws, float2* __restrict__ out)
{
    const int control = pctrl[0];
    const int nq      = pnq[0];
    const int mask    = 1 << (nq - control - 1);
    const size_t P    = (size_t)BATCH * DIM;
    const bf16_t* aplane[4] = { ws, ws + P, ws + 2 * P, ws + 3 * P };

    // 6 planes: 0=re_hi 1=re_lo 2=im_hi 3=im_lo 4=-im_hi 5=-im_lo
    // transposed [n][k], row stride 40 bf16 = 80 B (16B aligned, no bank dup).
    __shared__ __align__(16) bf16_t ldsB[6][BN][40];

    const int tid   = threadIdx.x;
    const int lane  = tid & 31;
    const int wave  = tid >> 5;
    const int nbase = blockIdx.x * BN;

    const int ml  = lane & 15;          // A row-in-tile
    const int kh  = lane >> 4;          // A k-half selector
    const int nl  = lane & 15;          // B col-in-tile
    const int kb2 = (lane >> 4) * 8;    // B k-base

    v8f accRe[2][2], accIm[2][2];
#pragma unroll
    for (int i = 0; i < 2; ++i)
#pragma unroll
        for (int j = 0; j < 2; ++j) { accRe[i][j] = {}; accIm[i][j] = {}; }

    // B staging work split: each thread converts 4 complex elements.
    const int skrow = tid >> 3;         // k row 0..31
    const int sngrp = (tid & 7) << 2;   // n group 0,4,...,28

    for (int kc = 0; kc < DIM; kc += KCH) {
        // Chunks are mask-uniform when mask >= KCH; fully-masked => A is zero.
        if (mask >= KCH && (kc & mask) == 0) continue;

        // ---- stage B chunk -> 6 bf16 planes in LDS (transposed) ----
        {
            const float4* bp = (const float4*)(tmat + (size_t)(kc + skrow) * DIM
                                               + (nbase + sngrp));
            float4 q0 = bp[0];
            float4 q1 = bp[1];
            if (kc + KCH < DIM)
                __builtin_prefetch((const void*)(tmat + (size_t)(kc + KCH + skrow) * DIM
                                                 + (nbase + sngrp)), 0, 0);
            float res[4] = { q0.x, q0.z, q1.x, q1.z };
            float ims[4] = { q0.y, q0.w, q1.y, q1.w };
#pragma unroll
            for (int j = 0; j < 4; ++j) {
                float re = res[j], im = ims[j];
                bf16_t rh  = (bf16_t)re;
                bf16_t rl  = (bf16_t)(re - (float)rh);
                bf16_t ih  = (bf16_t)im;
                bf16_t il  = (bf16_t)(im - (float)ih);
                bf16_t nih = (bf16_t)(-im);
                bf16_t nil = (bf16_t)(-(im - (float)ih));
                int n = sngrp + j;
                ldsB[0][n][skrow] = rh;
                ldsB[1][n][skrow] = rl;
                ldsB[2][n][skrow] = ih;
                ldsB[3][n][skrow] = il;
                ldsB[4][n][skrow] = nih;
                ldsB[5][n][skrow] = nil;
            }
        }
        __syncthreads();

        // ---- A fragments: direct from prepped planes (L2 resident) ----
        // ISA 16-bit A 16x32 layout: lane<16 -> K[0..7],K[16..23];
        // lane>=16 -> K[8..15],K[24..31]; two b128 loads per fragment.
        v16bf Af[2][4];
#pragma unroll
        for (int i = 0; i < 2; ++i) {
            int row = (wave * 2 + i) * 16 + ml;
            size_t off = (size_t)row * DIM + kc + 8 * kh;
#pragma unroll
            for (int p = 0; p < 4; ++p) {
                const int4* q = (const int4*)(aplane[p] + off);
                Af[i][p] = make_frag(q[0], q[2]);
            }
        }

        // ---- WMMA: 12 per complex C tile per K32 chunk ----
#pragma unroll
        for (int j = 0; j < 2; ++j) {
            v16bf Bf[6];
#pragma unroll
            for (int p = 0; p < 6; ++p) {
                const int4* q = (const int4*)&ldsB[p][j * 16 + nl][kb2];
                Bf[p] = make_frag(q[0], q[2]);
            }
#pragma unroll
            for (int i = 0; i < 2; ++i) {
                v16bf Rh = Af[i][0], Rl = Af[i][1], Ih = Af[i][2], Il = Af[i][3];
                v8f cr = accRe[i][j];
                cr = wmma_bf16(Rh, Bf[0], cr);   // re*re hi*hi
                cr = wmma_bf16(Rl, Bf[0], cr);   // lo*hi
                cr = wmma_bf16(Rh, Bf[1], cr);   // hi*lo
                cr = wmma_bf16(Ih, Bf[4], cr);   // -im*im hi*hi
                cr = wmma_bf16(Il, Bf[4], cr);
                cr = wmma_bf16(Ih, Bf[5], cr);
                accRe[i][j] = cr;
                v8f ci = accIm[i][j];
                ci = wmma_bf16(Rh, Bf[2], ci);   // re*im
                ci = wmma_bf16(Rl, Bf[2], ci);
                ci = wmma_bf16(Rh, Bf[3], ci);
                ci = wmma_bf16(Ih, Bf[0], ci);   // im*re
                ci = wmma_bf16(Il, Bf[0], ci);
                ci = wmma_bf16(Ih, Bf[1], ci);
                accIm[i][j] = ci;
            }
        }
        __syncthreads();
    }

    // ---- epilogue: control-bit select + store ----
    // C layout: vgpr r holds M=r (lanes 0-15) / M=r+8 (lanes 16-31); N = lane&15.
    const int rsel = (lane >> 4) * 8;
#pragma unroll
    for (int i = 0; i < 2; ++i) {
        int mt = wave * 2 + i;
#pragma unroll
        for (int j = 0; j < 2; ++j) {
            int n = nbase + j * 16 + (lane & 15);
            bool keep = (n & mask) != 0;
#pragma unroll
            for (int r = 0; r < 8; ++r) {
                int m = mt * 16 + rsel + r;
                float2 v;
                if (keep) { v.x = accRe[i][j][r]; v.y = accIm[i][j][r]; }
                else      { v = state[(size_t)m * DIM + n]; }
                out[(size_t)m * DIM + n] = v;
            }
        }
    }
}

extern "C" void kernel_launch(void* const* d_in, const int* in_sizes, int n_in,
                              void* d_out, int out_size, void* d_ws, size_t ws_size,
                              hipStream_t stream) {
    (void)in_sizes; (void)n_in; (void)out_size; (void)ws_size;
    const float2* state = (const float2*)d_in[0];
    const float2* tmat  = (const float2*)d_in[1];
    const int*    pctrl = (const int*)d_in[2];
    const int*    pnq   = (const int*)d_in[3];
    float2*       out   = (float2*)d_out;
    bf16_t*       ws    = (bf16_t*)d_ws;   // needs 4*256*8192*2 B = 16 MB

    dim3 pgrid((BATCH * DIM) / (NTHREADS * 8));   // 1024 blocks
    prep_a<<<pgrid, NTHREADS, 0, stream>>>(state, pctrl, pnq, ws);

    dim3 ggrid(DIM / BN);                          // 256 blocks
    cgemm_ctrl<<<ggrid, NTHREADS, 0, stream>>>(state, tmat, pctrl, pnq, ws, out);
}
